// MHAPyTorchScaledDotProduct_54863912239586
// MI455X (gfx1250) — matrix operand order repaired and checked by reading, not compiled, not run on hardware
//
#include <hip/hip_runtime.h>
#include <hip/hip_bf16.h>
#include <math.h>

// MHA forward: B=2, S=2048, D=1024, H=16, HD=64. f16 WMMA, f32 accumulate.

typedef __attribute__((ext_vector_type(16))) _Float16 v16h;
typedef __attribute__((ext_vector_type(8)))  _Float16 v8h;
typedef __attribute__((ext_vector_type(4)))  _Float16 v4h;
typedef __attribute__((ext_vector_type(8)))  float    v8f;

#define S_LEN  2048
#define DMODEL 1024
#define NHEAD  16
#define HDIM   64

__device__ __forceinline__ v8f wmma_f16(v16h a, v16h b, v8f c) {
  // D = A(16x32 f16) * B(32x16 f16) + C(16x16 f32)
  return __builtin_amdgcn_wmma_f32_16x16x32_f16(
      /*neg_a=*/false, a, /*neg_b=*/false, b,
      /*c_mod=*/(short)0, c, /*reuse_a=*/false, /*reuse_b=*/false);
}

// A-operand fragment: lane holds row M=lane%16; element e -> K = 16*g + e.
// => one contiguous 32B chunk at kbase+16g.
__device__ __forceinline__ v16h load_a_frag(const _Float16* __restrict__ row, int kbase, int g) {
  return *(const v16h*)(row + kbase + 16 * g);
}

// B-operand fragment: lane holds col N=lane%16; element e -> K = 16*g + e.
// => one contiguous 32B chunk at kbase+16g.
__device__ __forceinline__ v16h load_b_frag(const _Float16* __restrict__ row, int kbase, int g) {
  return *(const v16h*)(row + kbase + 16 * g);
}

// ---------------------------------------------------------------- converts ---
__global__ void convert_x_f16(const float4* __restrict__ x, v4h* __restrict__ xh) {
  int id = blockIdx.x * blockDim.x + threadIdx.x;
  float4 v = x[id];
  v4h o;
  o[0] = (_Float16)v.x; o[1] = (_Float16)v.y; o[2] = (_Float16)v.z; o[3] = (_Float16)v.w;
  xh[id] = o;
}

// W [1024 in][1024 out] f32 -> WT [out][in] f16 (LDS-tiled for coalescing)
__global__ void transpose_w_f16(const float* __restrict__ w, _Float16* __restrict__ wT) {
  __shared__ float tile[32][33];
  int i0 = blockIdx.x * 32;   // input-dim base
  int o0 = blockIdx.y * 32;   // output-dim base
  #pragma unroll
  for (int r = threadIdx.y; r < 32; r += 8)
    tile[r][threadIdx.x] = w[(size_t)(i0 + r) * DMODEL + o0 + threadIdx.x];
  __syncthreads();
  #pragma unroll
  for (int r = threadIdx.y; r < 32; r += 8)
    wT[(size_t)(o0 + r) * DMODEL + i0 + threadIdx.x] = (_Float16)tile[threadIdx.x][r];
}

// ------------------------------------------------------------------- GEMM ----
// C[M,N] = A[M,K] * Bt[N,K]^T.  Wave tile: 16(M) x 64(N).  Block: 8 waves = 32x256.
// MODE 0: store f16 row-major [M][1024]
// MODE 1: store f16 V-transposed: Vt[b][h][d][s]   (col = h*64+d, row = b*2048+s)
// MODE 2: store f32 row-major + bias
template <int MODE>
__global__ void gemm_wmma(const _Float16* __restrict__ A,
                          const _Float16* __restrict__ Bt,
                          void* __restrict__ outp,
                          const float* __restrict__ bias,
                          int K) {
  const int lane  = threadIdx.x & 31;
  const int wave  = threadIdx.x >> 5;
  const int n     = lane & 15;
  const int g     = lane >> 4;
  const int mbase = blockIdx.x * 32 + (wave >> 2) * 16;
  const int nbase = blockIdx.y * 256 + (wave & 3) * 64;

  const _Float16* Arow = A + (size_t)(mbase + n) * K;
  const _Float16* Br[4];
  #pragma unroll
  for (int t = 0; t < 4; ++t) Br[t] = Bt + (size_t)(nbase + 16 * t + n) * K;

  v8f acc[4] = {v8f{}, v8f{}, v8f{}, v8f{}};
  for (int k = 0; k < K; k += 32) {
    v16h a = load_a_frag(Arow, k, g);
    #pragma unroll
    for (int t = 0; t < 4; ++t)
      acc[t] = wmma_f16(a, load_b_frag(Br[t], k, g), acc[t]);
  }

  if (MODE == 0) {
    _Float16* O = (_Float16*)outp;
    #pragma unroll
    for (int t = 0; t < 4; ++t) {
      int col = nbase + 16 * t + n;
      #pragma unroll
      for (int r = 0; r < 8; ++r)
        O[(size_t)(mbase + 8 * g + r) * DMODEL + col] = (_Float16)acc[t][r];
    }
  } else if (MODE == 1) {
    _Float16* Vt = (_Float16*)outp;
    const int b  = mbase >> 11;            // S_LEN = 2048 rows per batch
    const int s0 = (mbase & 2047) + 8 * g; // 8 consecutive s per lane
    #pragma unroll
    for (int t = 0; t < 4; ++t) {
      int col = nbase + 16 * t + n;
      int h = col >> 6, d = col & 63;
      v8h o;
      #pragma unroll
      for (int r = 0; r < 8; ++r) o[r] = (_Float16)acc[t][r];
      *(v8h*)(Vt + ((size_t)((b * NHEAD + h) * HDIM + d)) * S_LEN + s0) = o;
    }
  } else {
    float* O = (float*)outp;
    #pragma unroll
    for (int t = 0; t < 4; ++t) {
      int col = nbase + 16 * t + n;
      float bv = bias[col];
      #pragma unroll
      for (int r = 0; r < 8; ++r)
        O[(size_t)(mbase + 8 * g + r) * DMODEL + col] = acc[t][r] + bv;
    }
  }
}

// ------------------------------------------------------------ flash attention
// One wave per (b, h, 16-query tile). Scores computed transposed: S^T = K * Q^T
// so probabilities carry the query index in the lane (matches B-operand and
// C/D layouts). Context accumulated as O^T = V^T * P^T with V stored [b][h][d][s].
__global__ void flash_attn(const _Float16* __restrict__ Qh,
                           const _Float16* __restrict__ Kh,
                           const _Float16* __restrict__ Vt,
                           _Float16* __restrict__ ctx) {
  const int lane = threadIdx.x & 31;
  const int wave = threadIdx.x >> 5;
  const int n = lane & 15;
  const int g = lane >> 4;

  const int w  = blockIdx.x * 8 + wave;
  const int qt = w & 127;          // S/16 tiles
  const int h  = (w >> 7) & 15;
  const int b  = w >> 11;
  const int qbase = qt * 16;

  // Q^T B-operand fragments (lane = query n), head dim split into 2 k-steps.
  const _Float16* Qrow = Qh + (size_t)(b * S_LEN + qbase + n) * DMODEL + h * HDIM;
  const v16h bq0 = load_b_frag(Qrow, 0, g);
  const v16h bq1 = load_b_frag(Qrow, 32, g);

  const _Float16* Kbase = Kh + (size_t)(b * S_LEN + n) * DMODEL + h * HDIM;
  const _Float16* Vr0   = Vt + (size_t)((b * NHEAD + h) * HDIM + n) * S_LEN;

  float m = -INFINITY, l = 0.f;
  v8f o0 = {}, o1 = {}, o2 = {}, o3 = {};

  for (int kb = 0; kb < S_LEN; kb += 32) {
    const _Float16* K0 = Kbase + (size_t)kb * DMODEL;
    const _Float16* K1 = K0 + (size_t)16 * DMODEL;

    // S^T tiles: t0 keys [kb,kb+16), t1 keys [kb+16,kb+32); C layout:
    // lane holds [key = r + 8g][query = n].
    v8f t0 = {}, t1 = {};
    t0 = wmma_f16(load_a_frag(K0, 0, g),  bq0, t0);
    t0 = wmma_f16(load_a_frag(K0, 32, g), bq1, t0);
    t1 = wmma_f16(load_a_frag(K1, 0, g),  bq0, t1);
    t1 = wmma_f16(load_a_frag(K1, 32, g), bq1, t1);

    float s0[8], s1[8];
    float mloc = -INFINITY;
    #pragma unroll
    for (int i = 0; i < 8; ++i) {
      s0[i] = t0[i] * 0.125f;   // 1/sqrt(HD)
      s1[i] = t1[i] * 0.125f;
      mloc = fmaxf(mloc, fmaxf(s0[i], s1[i]));
    }
    mloc = fmaxf(mloc, __shfl_xor(mloc, 16));   // other half of the keys
    const float mnew  = fmaxf(m, mloc);
    const float alpha = __expf(m - mnew);

    float p0[8], p1[8], ps = 0.f;
    #pragma unroll
    for (int i = 0; i < 8; ++i) {
      p0[i] = __expf(s0[i] - mnew);
      p1[i] = __expf(s1[i] - mnew);
      ps += p0[i] + p1[i];
    }
    l = l * alpha + ps;
    m = mnew;
    #pragma unroll
    for (int i = 0; i < 8; ++i) {
      o0[i] *= alpha; o1[i] *= alpha; o2[i] *= alpha; o3[i] *= alpha;
    }

    // Build P^T B-operand (lane = query, element e -> key 16g+e).
    // g=0 lane holds keys 0..7 (p0) and needs partner's p0 (keys 8..15);
    // g=1 lane holds keys 24..31 (p1) and needs partner's p1 (keys 16..23).
    v16h bp;
    #pragma unroll
    for (int i = 0; i < 8; ++i) {
      float send = g ? p0[i] : p1[i];         // what the partner lane needs
      float rec  = __shfl_xor(send, 16);
      bp[i]     = (_Float16)(g ? rec : p0[i]);
      bp[8 + i] = (_Float16)(g ? p1[i] : rec);
    }

    // O^T += V^T (d rows in lanes) * P^T
    o0 = wmma_f16(load_a_frag(Vr0,                kb, g), bp, o0);
    o1 = wmma_f16(load_a_frag(Vr0 + 16 * S_LEN,   kb, g), bp, o1);
    o2 = wmma_f16(load_a_frag(Vr0 + 32 * S_LEN,   kb, g), bp, o2);
    o3 = wmma_f16(load_a_frag(Vr0 + 48 * S_LEN,   kb, g), bp, o3);
  }

  const float ltot = l + __shfl_xor(l, 16);
  const float inv  = 1.f / ltot;

  // O^T C layout: lane holds [d = 16t + r + 8g][query n] -> 16B packed stores.
  _Float16* Crow = ctx + (size_t)(b * S_LEN + qbase + n) * DMODEL + h * HDIM + 8 * g;
  v8f oo[4] = {o0, o1, o2, o3};
  #pragma unroll
  for (int t = 0; t < 4; ++t) {
    v8h ov;
    #pragma unroll
    for (int r = 0; r < 8; ++r) ov[r] = (_Float16)(oo[t][r] * inv);
    *(v8h*)(Crow + 16 * t) = ov;
  }
}

// ---------------------------------------------------------------- launcher ---
extern "C" void kernel_launch(void* const* d_in, const int* in_sizes, int n_in,
                              void* d_out, int out_size, void* d_ws, size_t ws_size,
                              hipStream_t stream) {
  const float* x  = (const float*)d_in[0];
  const float* wq = (const float*)d_in[1];
  const float* wk = (const float*)d_in[2];
  const float* wv = (const float*)d_in[3];
  const float* wo = (const float*)d_in[4];
  const float* bo = (const float*)d_in[5];
  float* out = (float*)d_out;

  // Workspace layout (element = _Float16); total 48 MB.
  _Float16* ws  = (_Float16*)d_ws;
  _Float16* xh  = ws;             // 4096x1024
  _Float16* wqT = ws + 4194304;   // 1024x1024 (transposed)
  _Float16* wkT = ws + 5242880;
  _Float16* wvT = ws + 6291456;
  _Float16* woT = ws + 7340032;
  _Float16* Qh  = ws + 8388608;   // [b][s][h*64+d]
  _Float16* Kh  = ws + 12582912;  // [b][s][h*64+d]
  _Float16* Vt  = ws + 16777216;  // [b][h][d][s]
  _Float16* ctx = ws + 20971520;  // [b][s][h*64+d]

  convert_x_f16<<<4096, 256, 0, stream>>>((const float4*)x, (v4h*)xh);

  dim3 tb(32, 8), tg(32, 32);
  transpose_w_f16<<<tg, tb, 0, stream>>>(wq, wqT);
  transpose_w_f16<<<tg, tb, 0, stream>>>(wk, wkT);
  transpose_w_f16<<<tg, tb, 0, stream>>>(wv, wvT);
  transpose_w_f16<<<tg, tb, 0, stream>>>(wo, woT);

  dim3 gg(128, 4);  // 4096/32 x 1024/256
  gemm_wmma<0><<<gg, 256, 0, stream>>>(xh, wqT, Qh, nullptr, 1024);
  gemm_wmma<0><<<gg, 256, 0, stream>>>(xh, wkT, Kh, nullptr, 1024);
  gemm_wmma<1><<<gg, 256, 0, stream>>>(xh, wvT, Vt, nullptr, 1024);

  flash_attn<<<512, 256, 0, stream>>>(Qh, Kh, Vt, ctx);  // 4096 waves

  gemm_wmma<2><<<gg, 256, 0, stream>>>(ctx, woT, out, bo, 1024);
}
